// MyLongformer_2104533975736
// MI455X (gfx1250) — compile-verified, matrix operand order checked by reading, not run on hardware
//
#include <hip/hip_runtime.h>
#include <hip/hip_bf16.h>
#include <math.h>

// ---------------- model constants ----------------
constexpr int kS  = 2048;
constexpr int kD  = 768;
constexpr int kH  = 12;
constexpr int kHD = 64;
constexpr int kF  = 3072;
constexpr int kC  = 256;
constexpr int kG  = 16;
constexpr int kL  = 12;
constexpr int kNC = kS / kC;          // 8 chunks
constexpr int kKeys = kG + 3 * kC;    // 784 (global + window)
constexpr float kNEG = -1.0e9f;
constexpr float kSCALE = 0.125f;      // 1/sqrt(64)

typedef __bf16 bf16;
typedef __attribute__((ext_vector_type(16))) __bf16 v16bf;
typedef __attribute__((ext_vector_type(8)))  float  v8f;

// ---------------- WMMA fragment load (16-bit A/B layout, CDNA5 ISA 7.12.2) ----
// LDS tile is [rows][32] bf16.  Lane l holds row r = base + (l&15); half = l>>4.
// VGPR j holds the bf16 pair at K = (j<4?0:16) + half*8 + (j&3)*2  -> dword
// index in the 32-element row = (j<4?0:8) + half*4 + (j&3).
union FragU { v16bf v; unsigned u[8]; };

__device__ inline v16bf load_frag16(const bf16* tile, int row_base, int lane) {
    int r    = row_base + (lane & 15);
    int half = lane >> 4;
    const unsigned* p = (const unsigned*)(tile + r * 32);
    FragU f;
#pragma unroll
    for (int j = 0; j < 8; ++j) {
        int idx = ((j < 4) ? 0 : 8) + half * 4 + (j & 3);
        f.u[j] = p[idx];
    }
    return f.v;
}

// ---------------- block reduction helper (256 threads) ----------------
__device__ inline float block_reduce_sum256(float v, float* red) {
    int t = threadIdx.x;
    red[t] = v;
    __syncthreads();
#pragma unroll
    for (int s = 128; s > 0; s >>= 1) {
        if (t < s) red[t] += red[t + s];
        __syncthreads();
    }
    float r = red[0];
    __syncthreads();
    return r;
}

// =====================================================================
// GEMM: C[M,N] = act( (A_bf16[M,K] x W_f32->bf16[K,N] + bias) * scale )
// 128x128 tile, K-step 32, 8 waves -> each wave a 32x64 sub-tile
// (2x4 accumulators of v_wmma_f32_16x16x32_bf16).
// A tile staged via gfx1250 async DMA (global_load_async_to_lds_b128,
// ASYNCcnt); W tile staged through VGPRs for the fp32->bf16 convert.
// =====================================================================
__global__ __launch_bounds__(256)
void gemm_bf16_kernel(const bf16* __restrict__ A, const float* __restrict__ W,
                      const float* __restrict__ bias,
                      float* __restrict__ Cout, bf16* __restrict__ Cbf,
                      int M, int N, int K, float scale, int act)
{
    __shared__ __align__(16) bf16 As[128 * 32];
    __shared__ __align__(16) bf16 Bs[128 * 32];   // stored transposed: [n][k]

    const int tid  = threadIdx.x;
    const int lane = tid & 31;
    const int wid  = tid >> 5;
    const int wm   = wid >> 1;        // 0..3 -> 32-row band
    const int wn   = wid & 1;         // 0..1 -> 64-col band
    const int m0   = blockIdx.y * 128;
    const int n0   = blockIdx.x * 128;

    v8f acc[2][4];
#pragma unroll
    for (int i = 0; i < 2; ++i)
#pragma unroll
        for (int j = 0; j < 4; ++j) {
            v8f z = {0.f,0.f,0.f,0.f,0.f,0.f,0.f,0.f};
            acc[i][j] = z;
        }

    // per-thread staging coordinates (invariant across K loop)
    const int ar    = tid >> 1;            // A row within tile 0..127
    const int akofs = (tid & 1) * 16;      // A col offset 0 or 16
    const int agmr  = m0 + ar;             // global A row
    bf16* adst = &As[ar * 32 + akofs];
    const unsigned ldsA = (unsigned)(unsigned long long)adst;   // LDS byte addr

    const int kr = tid >> 3;               // W row within K-slice 0..31
    const int cg = (tid & 7) * 16;         // W col group 0,16,...,112

    for (int kt = 0; kt < K; kt += 32) {
        // --- stage A tile: 128x32 bf16 via async DMA straight into LDS ---
        if (agmr < M) {
            unsigned long long ga =
                (unsigned long long)(A + (size_t)agmr * K + kt + akofs);
            asm volatile("global_load_async_to_lds_b128 %0, %1, off"
                         :: "v"(ldsA), "v"(ga) : "memory");
            asm volatile("global_load_async_to_lds_b128 %0, %1, off offset:16"
                         :: "v"(ldsA), "v"(ga) : "memory");
        } else {
#pragma unroll
            for (int i = 0; i < 16; ++i) adst[i] = (bf16)0.f;
        }
        // --- stage W tile: fp32 [kt..kt+31][n0..n0+127] -> bf16, transposed ---
        {
            const float* src = W + (size_t)(kt + kr) * N + n0 + cg;
            if (kt + 32 < K)                       // pull next slice toward L2
                __builtin_prefetch(src + (size_t)32 * N, 0, 1);
#pragma unroll
            for (int i = 0; i < 16; i += 4) {
                float4 f = *(const float4*)(src + i);
                Bs[(cg + i + 0) * 32 + kr] = (bf16)f.x;
                Bs[(cg + i + 1) * 32 + kr] = (bf16)f.y;
                Bs[(cg + i + 2) * 32 + kr] = (bf16)f.z;
                Bs[(cg + i + 3) * 32 + kr] = (bf16)f.w;
            }
        }
        // async LDS writes must be complete before the barrier publishes them
        asm volatile("s_wait_asynccnt 0x0" ::: "memory");
        __syncthreads();

        v16bf a0 = load_frag16(As, wm * 32,      lane);
        v16bf a1 = load_frag16(As, wm * 32 + 16, lane);
#pragma unroll
        for (int sn = 0; sn < 4; ++sn) {
            v16bf b = load_frag16(Bs, wn * 64 + sn * 16, lane);
            acc[0][sn] = __builtin_amdgcn_wmma_f32_16x16x32_bf16(
                false, a0, false, b, (short)0, acc[0][sn], false, false);
            acc[1][sn] = __builtin_amdgcn_wmma_f32_16x16x32_bf16(
                false, a1, false, b, (short)0, acc[1][sn], false, false);
        }
        __syncthreads();
    }

    // --- epilogue: bias + scale + optional exact GELU; fp32 and/or bf16 out ---
#pragma unroll
    for (int sm = 0; sm < 2; ++sm) {
#pragma unroll
        for (int sn = 0; sn < 4; ++sn) {
            int n = n0 + wn * 64 + sn * 16 + (lane & 15);
            float bval = (bias && n < N) ? bias[n] : 0.f;
#pragma unroll
            for (int i = 0; i < 8; ++i) {
                int m = m0 + wm * 32 + sm * 16 + (lane >> 4) * 8 + i;
                if (m < M && n < N) {
                    float v = (acc[sm][sn][i] + bval) * scale;
                    if (act == 1) v = 0.5f * v * (1.f + erff(v * 0.70710678118f));
                    if (Cout) Cout[(size_t)m * N + n] = v;
                    if (Cbf)  Cbf[(size_t)m * N + n] = (bf16)v;
                }
            }
        }
    }
}

// =====================================================================
// Embedding gather + LayerNorm  (one block per token; D=768 -> 3/thread)
// =====================================================================
__global__ __launch_bounds__(256)
void embed_ln_kernel(const float* __restrict__ word_emb,
                     const float* __restrict__ pos_emb,
                     const int* __restrict__ ids,
                     const float* __restrict__ g, const float* __restrict__ b,
                     float* __restrict__ h, bf16* __restrict__ hb)
{
    __shared__ float red[256];
    int s  = blockIdx.x;
    int id = ids[s];
    float x[3];
    float psum = 0.f;
#pragma unroll
    for (int i = 0; i < 3; ++i) {
        int d = threadIdx.x + i * 256;
        x[i] = word_emb[(size_t)id * kD + d] + pos_emb[(size_t)s * kD + d];
        psum += x[i];
    }
    float mu = block_reduce_sum256(psum, red) * (1.f / kD);
    float pv = 0.f;
#pragma unroll
    for (int i = 0; i < 3; ++i) { float t = x[i] - mu; pv += t * t; }
    float var = block_reduce_sum256(pv, red) * (1.f / kD);
    float inv = rsqrtf(var + 1e-5f);
#pragma unroll
    for (int i = 0; i < 3; ++i) {
        int d = threadIdx.x + i * 256;
        float y = (x[i] - mu) * inv * g[d] + b[d];
        h[(size_t)s * kD + d]  = y;
        hb[(size_t)s * kD + d] = (bf16)y;
    }
}

// =====================================================================
// Residual add + LayerNorm (in place on h; also emits bf16)
// =====================================================================
__global__ __launch_bounds__(256)
void add_ln_kernel(float* __restrict__ h, const float* __restrict__ delta,
                   const float* __restrict__ g, const float* __restrict__ b,
                   bf16* __restrict__ hb)
{
    __shared__ float red[256];
    int s = blockIdx.x;
    float x[3];
    float psum = 0.f;
#pragma unroll
    for (int i = 0; i < 3; ++i) {
        int d = threadIdx.x + i * 256;
        x[i] = h[(size_t)s * kD + d] + delta[(size_t)s * kD + d];
        psum += x[i];
    }
    float mu = block_reduce_sum256(psum, red) * (1.f / kD);
    float pv = 0.f;
#pragma unroll
    for (int i = 0; i < 3; ++i) { float t = x[i] - mu; pv += t * t; }
    float var = block_reduce_sum256(pv, red) * (1.f / kD);
    float inv = rsqrtf(var + 1e-5f);
#pragma unroll
    for (int i = 0; i < 3; ++i) {
        int d = threadIdx.x + i * 256;
        float y = (x[i] - mu) * inv * g[d] + b[d];
        h[(size_t)s * kD + d]  = y;
        hb[(size_t)s * kD + d] = (bf16)y;
    }
}

// =====================================================================
// Local (windowed) attention + global-key columns.  One wave per query.
// grid = (C/8, NC, H);  scores: [16 global | 768 windowed] in LDS.
// =====================================================================
__global__ __launch_bounds__(256)
void attn_local_kernel(const float* __restrict__ q, const float* __restrict__ k,
                       const float* __restrict__ v,
                       const int* __restrict__ am, const int* __restrict__ gm,
                       bf16* __restrict__ outb)
{
    __shared__ float sc[8][kKeys];
    __shared__ __align__(16) float qv[8][kHD];

    int w    = threadIdx.x >> 5;
    int lane = threadIdx.x & 31;
    int h    = blockIdx.z;
    int n    = blockIdx.y;
    int qc   = blockIdx.x * 8 + w;      // query index within chunk 0..255
    int s    = n * kC + qc;             // global query index

    const float* qrow = q + (size_t)s * kD + h * kHD;
    qv[w][lane]      = qrow[lane];
    qv[w][lane + 32] = qrow[lane + 32];
    __syncthreads();

    // ---- scores ----
    for (int j = lane; j < kKeys; j += 32) {
        float sv = kNEG;
        int kk; bool valid;
        if (j < kG) {                        // global-key columns use k[:, :G]
            kk = j;
            valid = (am[kk] > 0);
        } else {
            int jj = j - kG;                 // 0..767 window position
            kk = (n - 1) * kC + jj;
            int rel = jj - kC - qc;
            valid = (kk >= 0) && (kk < kS) && (rel >= -kC) && (rel <= kC)
                    && (am[kk] > 0) && (gm[kk] == 0);
        }
        if (valid) {
            const float4* kp = (const float4*)(k + (size_t)kk * kD + h * kHD);
            const float4* qp = (const float4*)(qv[w]);
            float acc = 0.f;
#pragma unroll
            for (int d4 = 0; d4 < 16; ++d4) {
                float4 a = qp[d4], bb = kp[d4];
                acc += a.x*bb.x + a.y*bb.y + a.z*bb.z + a.w*bb.w;
            }
            sv = acc;
        }
        sc[w][j] = sv;
    }

    // ---- softmax (wave-level) ----
    float mx = kNEG;
    for (int j = lane; j < kKeys; j += 32) mx = fmaxf(mx, sc[w][j]);
#pragma unroll
    for (int o = 16; o > 0; o >>= 1) mx = fmaxf(mx, __shfl_xor(mx, o, 32));
    float sum = 0.f;
    for (int j = lane; j < kKeys; j += 32) {
        float e = __expf(sc[w][j] - mx);
        sc[w][j] = e;
        sum += e;
    }
#pragma unroll
    for (int o = 16; o > 0; o >>= 1) sum += __shfl_xor(sum, o, 32);
    float inv = 1.f / sum;
    __syncthreads();

    // ---- weighted sum of values (lane owns 2 of 64 dims) ----
    int d0 = lane * 2;
    float o0 = 0.f, o1 = 0.f;
    for (int j = 0; j < kKeys; ++j) {
        int kk;
        if (j < kG) kk = j;
        else {
            kk = (n - 1) * kC + (j - kG);
            if (kk < 0 || kk >= kS) continue;    // padded window position
        }
        float p = sc[w][j];
        const float* vp = v + (size_t)kk * kD + h * kHD;
        o0 += p * vp[d0];
        o1 += p * vp[d0 + 1];
    }
    outb[(size_t)s * kD + h * kHD + d0]     = (bf16)(o0 * inv);
    outb[(size_t)s * kD + h * kHD + d0 + 1] = (bf16)(o1 * inv);
}

// =====================================================================
// Global attention for the G=16 global queries over all S keys.
// Flash-style online softmax, one wave per (query, head).  grid = 2*H.
// =====================================================================
__global__ __launch_bounds__(256)
void attn_global_kernel(const float* __restrict__ qg, const float* __restrict__ kg,
                        const float* __restrict__ vg, const int* __restrict__ am,
                        bf16* __restrict__ outb)
{
    __shared__ __align__(16) float qv[8][kHD];
    int w    = threadIdx.x >> 5;
    int lane = threadIdx.x & 31;
    int h    = blockIdx.x >> 1;
    int gq   = (blockIdx.x & 1) * 8 + w;     // 0..15

    const float* qrow = qg + (size_t)gq * kD + h * kHD;
    qv[w][lane]      = qrow[lane];
    qv[w][lane + 32] = qrow[lane + 32];
    __syncthreads();

    float m = kNEG, l = 0.f;
    int d0 = lane * 2;
    float o0 = 0.f, o1 = 0.f;

    for (int kb = 0; kb < kS; kb += 32) {
        int kk = kb + lane;
        float sv = kNEG;
        if (am[kk] > 0) {
            const float4* kp = (const float4*)(kg + (size_t)kk * kD + h * kHD);
            const float4* qp = (const float4*)(qv[w]);
            float acc = 0.f;
#pragma unroll
            for (int d4 = 0; d4 < 16; ++d4) {
                float4 a = qp[d4], bb = kp[d4];
                acc += a.x*bb.x + a.y*bb.y + a.z*bb.z + a.w*bb.w;
            }
            sv = acc;
        }
        float bm = sv;
#pragma unroll
        for (int o = 16; o > 0; o >>= 1) bm = fmaxf(bm, __shfl_xor(bm, o, 32));
        float nm = fmaxf(m, bm);
        float rescale = __expf(m - nm);
        float e = __expf(sv - nm);
        float bs = e;
#pragma unroll
        for (int o = 16; o > 0; o >>= 1) bs += __shfl_xor(bs, o, 32);
        l = l * rescale + bs;
        o0 *= rescale; o1 *= rescale;
        m = nm;
#pragma unroll 4
        for (int t = 0; t < 32; ++t) {
            float p = __shfl(e, t, 32);
            const float* vp = vg + (size_t)(kb + t) * kD + h * kHD;
            o0 += p * vp[d0];
            o1 += p * vp[d0 + 1];
        }
    }
    float inv = 1.f / l;
    outb[(size_t)gq * kD + h * kHD + d0]     = (bf16)(o0 * inv);
    outb[(size_t)gq * kD + h * kHD + d0 + 1] = (bf16)(o1 * inv);
}

// =====================================================================
// pooled = tanh(h[0] @ pool_w + pool_b)    (768 outputs)
// =====================================================================
__global__ __launch_bounds__(256)
void pool_kernel(const float* __restrict__ h, const float* __restrict__ pool_w,
                 const float* __restrict__ pool_b, float* __restrict__ pooled)
{
    int j = blockIdx.x * 256 + threadIdx.x;
    if (j < kD) {
        float acc = 0.f;
        for (int d = 0; d < kD; ++d) acc += h[d] * pool_w[(size_t)d * kD + j];
        pooled[j] = tanhf(acc + pool_b[j]);
    }
}

// =====================================================================
// prob = sigmoid(pooled @ head_w + head_b); BCE loss  -> d_out[0..1]
// =====================================================================
__global__ __launch_bounds__(256)
void head_loss_kernel(const float* __restrict__ pooled,
                      const float* __restrict__ head_w,
                      const float* __restrict__ head_b,
                      const float* __restrict__ labels,
                      float* __restrict__ out)
{
    __shared__ float red[256];
    float a = 0.f;
    for (int d = threadIdx.x; d < kD; d += 256) a += pooled[d] * head_w[d];
    float z = block_reduce_sum256(a, red);
    if (threadIdx.x == 0) {
        z += head_b[0];
        float prob = 1.f / (1.f + __expf(-z));
        float p = fminf(fmaxf(prob, 1e-7f), 1.f - 1e-7f);
        float lab = labels[0];
        float loss = -(lab * logf(p) + (1.f - lab) * log1pf(-p));
        out[0] = prob;
        out[1] = loss;
    }
}

// =====================================================================
// host orchestration
// =====================================================================
extern "C" void kernel_launch(void* const* d_in, const int* in_sizes, int n_in,
                              void* d_out, int out_size, void* d_ws, size_t ws_size,
                              hipStream_t stream)
{
    const float* word_emb = (const float*)d_in[0];
    const float* pos_emb  = (const float*)d_in[1];
    const float* ln_e_g   = (const float*)d_in[2];
    const float* ln_e_b   = (const float*)d_in[3];
    const float* wq  = (const float*)d_in[4];
    const float* bq  = (const float*)d_in[5];
    const float* wk  = (const float*)d_in[6];
    const float* bk  = (const float*)d_in[7];
    const float* wv  = (const float*)d_in[8];
    const float* bv  = (const float*)d_in[9];
    const float* wqg = (const float*)d_in[10];
    const float* bqg = (const float*)d_in[11];
    const float* wkg = (const float*)d_in[12];
    const float* bkg = (const float*)d_in[13];
    const float* wvg = (const float*)d_in[14];
    const float* bvg = (const float*)d_in[15];
    const float* wo  = (const float*)d_in[16];
    const float* bo  = (const float*)d_in[17];
    const float* ln1_g = (const float*)d_in[18];
    const float* ln1_b = (const float*)d_in[19];
    const float* w1 = (const float*)d_in[20];
    const float* b1 = (const float*)d_in[21];
    const float* w2 = (const float*)d_in[22];
    const float* b2 = (const float*)d_in[23];
    const float* ln2_g = (const float*)d_in[24];
    const float* ln2_b = (const float*)d_in[25];
    const float* pool_w = (const float*)d_in[26];
    const float* pool_b = (const float*)d_in[27];
    const float* head_w = (const float*)d_in[28];
    const float* head_b = (const float*)d_in[29];
    const float* labels = (const float*)d_in[30];
    const int* input_ids = (const int*)d_in[31];
    const int* am        = (const int*)d_in[32];
    const int* gm        = (const int*)d_in[33];

    const size_t SD = (size_t)kS * kD;    // 1.57 M
    const size_t SF = (size_t)kS * kF;    // 6.29 M

    // bump allocator in workspace (256B aligned)
    char* p = (char*)d_ws;
    auto alloc = [&](size_t bytes) -> void* {
        void* r = (void*)p;
        p += (bytes + 255) & ~((size_t)255);
        return r;
    };
    float* h      = (float*)alloc(SD * 4);
    float* q      = (float*)alloc(SD * 4);
    float* k      = (float*)alloc(SD * 4);
    float* v      = (float*)alloc(SD * 4);
    float* kgb    = (float*)alloc(SD * 4);
    float* vgb    = (float*)alloc(SD * 4);
    float* qgb    = (float*)alloc((size_t)kG * kD * 4);
    float* proj   = (float*)alloc(SD * 4);
    float* pooled = (float*)alloc(kD * 4);
    bf16*  hb     = (bf16*)alloc(SD * 2);
    bf16*  attnb  = (bf16*)alloc(SD * 2);
    bf16*  f1b    = (bf16*)alloc(SF * 2);

    auto gemm = [&](const bf16* A, const float* W, const float* Bias,
                    float* Cout, bf16* Cbf, int M, int N, int K,
                    float scale, int act) {
        dim3 grid((N + 127) / 128, (M + 127) / 128);
        gemm_bf16_kernel<<<grid, 256, 0, stream>>>(A, W, Bias, Cout, Cbf,
                                                   M, N, K, scale, act);
    };

    // embedding + LN
    embed_ln_kernel<<<kS, 256, 0, stream>>>(word_emb, pos_emb, input_ids,
                                            ln_e_g, ln_e_b, h, hb);

    const size_t DD = (size_t)kD * kD;
    const size_t DF = (size_t)kD * kF;

    for (int l = 0; l < kL; ++l) {
        // projections (bf16 WMMA GEMMs); q/qg get the 1/sqrt(HD) scale fused
        gemm(hb, wq  + l*DD, bq  + (size_t)l*kD, q,   nullptr, kS, kD, kD, kSCALE, 0);
        gemm(hb, wk  + l*DD, bk  + (size_t)l*kD, k,   nullptr, kS, kD, kD, 1.f, 0);
        gemm(hb, wv  + l*DD, bv  + (size_t)l*kD, v,   nullptr, kS, kD, kD, 1.f, 0);
        gemm(hb, wkg + l*DD, bkg + (size_t)l*kD, kgb, nullptr, kS, kD, kD, 1.f, 0);
        gemm(hb, wvg + l*DD, bvg + (size_t)l*kD, vgb, nullptr, kS, kD, kD, 1.f, 0);
        gemm(hb, wqg + l*DD, bqg + (size_t)l*kD, qgb, nullptr, kG, kD, kD, kSCALE, 0);

        // attention
        attn_local_kernel<<<dim3(kC / 8, kNC, kH), 256, 0, stream>>>(
            q, k, v, am, gm, attnb);
        attn_global_kernel<<<2 * kH, 256, 0, stream>>>(qgb, kgb, vgb, am, attnb);

        // output projection + residual LN
        gemm(attnb, wo + l*DD, bo + (size_t)l*kD, proj, nullptr, kS, kD, kD, 1.f, 0);
        add_ln_kernel<<<kS, 256, 0, stream>>>(h, proj, ln1_g + (size_t)l*kD,
                                              ln1_b + (size_t)l*kD, hb);

        // FFN: GEMM1 with fused exact GELU (bf16 out), GEMM2, residual LN
        gemm(hb, w1 + l*DF, b1 + (size_t)l*kF, nullptr, f1b, kS, kF, kD, 1.f, 1);
        gemm(f1b, w2 + l*DF, b2 + (size_t)l*kD, proj, nullptr, kS, kD, kF, 1.f, 0);
        add_ln_kernel<<<kS, 256, 0, stream>>>(h, proj, ln2_g + (size_t)l*kD,
                                              ln2_b + (size_t)l*kD, hb);
    }

    // pooled head + loss
    pool_kernel<<<(kD + 255) / 256, 256, 0, stream>>>(h, pool_w, pool_b, pooled);
    head_loss_kernel<<<1, 256, 0, stream>>>(pooled, head_w, head_b, labels,
                                            (float*)d_out);
}
